// sp_attn_head_l0_26963804684999
// MI455X (gfx1250) — matrix operand order repaired and checked by reading, compiled, not run on hardware
//
#include <hip/hip_runtime.h>
#include <hip/hip_fp16.h>

typedef __attribute__((ext_vector_type(16))) _Float16 v16h;
typedef __attribute__((ext_vector_type(8)))  float    v8f;

#define F_IN 256
#define D_OUT 128
#define ALPHA 0.2f

// ---------------------------------------------------------------- utilities
__device__ __forceinline__ void atomicMaxFloat(float* addr, float val) {
    // Exact float max via signed/unsigned integer atomic ordering trick.
    if (val >= 0.0f)
        atomicMax((int*)addr, __float_as_int(val));
    else
        atomicMin((unsigned int*)addr, __float_as_uint(val));
}

// ---------------------------------------------------------------- kernel 0: init
// Zero the output accumulator, set segment-max to -inf, denom to 0.
__global__ __launch_bounds__(256) void gat_init_kernel(float* __restrict__ out,
                                                       float* __restrict__ m,
                                                       float* __restrict__ denom,
                                                       int n) {
    int i = blockIdx.x * blockDim.x + threadIdx.x;
    int total = n * D_OUT;
    if (i < total) out[i] = 0.0f;
    if (i < n) {
        m[i] = -__builtin_inff();
        denom[i] = 0.0f;
    }
}

// ---------------------------------------------------------------- kernel 1: GEMM h = seq @ W
// Block: 256 threads = 8 waves. Block computes a 16-row x 128-col tile of h.
// Each wave owns one 16x16 output tile; K loop = 8 steps of v_wmma_f32_16x16x32_f16.
__global__ __launch_bounds__(256) void gat_gemm_kernel(const float* __restrict__ seq,
                                                       const float* __restrict__ W,
                                                       float* __restrict__ h,
                                                       int n) {
    __shared__ _Float16 As[16 * F_IN];   // 16-row A tile, converted to f16 (8 KB)

    const int tid  = threadIdx.x;
    const int wave = tid >> 5;
    const int lane = tid & 31;
    const int row0 = blockIdx.x * 16;

    // Cooperative load of the 16x256 A tile into LDS (f32 -> f16).
    {
        int base = tid * 16;                 // 256 threads * 16 elems = 4096
        int r    = base >> 8;                // /256
        int kk   = base & (F_IN - 1);
        int arow = row0 + r;
        if (arow >= n) arow = n - 1;         // clamp (N may not be tile-multiple)
        const float* src = seq + (size_t)arow * F_IN + kk;
#pragma unroll
        for (int i = 0; i < 16; ++i)
            As[base + i] = (_Float16)src[i];
    }
    __syncthreads();

    const int lm  = lane & 15;
    const int hi  = lane >> 4;       // 0: lanes 0-15, 1: lanes 16-31
    const int kbA = hi * 8;          // A: upper lanes hold K+8 block
    const int kbB = hi * 8;          // B: upper lanes hold K+8 block
    const int ncol = wave * 16 + lm; // output column (0..127)

    v8f acc = {};
#pragma unroll
    for (int k0 = 0; k0 < F_IN; k0 += 32) {
        v16h a, b;
        // A fragment: lane holds row lm; K = k0+kbA+i and k0+16+kbA+i
#pragma unroll
        for (int i = 0; i < 8; ++i) {
            a[i]     = As[lm * F_IN + k0 + kbA + i];
            a[8 + i] = As[lm * F_IN + k0 + 16 + kbA + i];
        }
        // B fragment: lane holds column ncol; K = k0+kbB+i and k0+16+kbB+i
#pragma unroll
        for (int i = 0; i < 16; ++i)
            b[i] = (_Float16)W[(size_t)(k0 + kbB + (i & 7) + ((i >> 3) << 4)) * D_OUT + ncol];

        acc = __builtin_amdgcn_wmma_f32_16x16x32_f16(
            /*neg_a=*/false, a, /*neg_b=*/false, b,
            /*c_mod=*/(short)0, acc, /*reuse_a=*/false, /*reuse_b=*/false);
    }

    // C/D layout: VGPR r holds M=r (lanes 0-15) or M=8+r (lanes 16-31), N=lane%16.
#pragma unroll
    for (int r = 0; r < 8; ++r) {
        int mrow = row0 + (hi ? 8 + r : r);
        if (mrow < n)
            h[(size_t)mrow * D_OUT + ncol] = acc[r];
    }
}

// ---------------------------------------------------------------- kernel 2: f1/f2 per node
// One wave per row; lane loads float4, wave32 shuffle reduction.
__global__ __launch_bounds__(256) void gat_f12_kernel(const float* __restrict__ h,
                                                      const float* __restrict__ a1w,
                                                      const float* __restrict__ a2w,
                                                      const float* __restrict__ a1b,
                                                      const float* __restrict__ a2b,
                                                      float* __restrict__ f1,
                                                      float* __restrict__ f2,
                                                      int n) {
    int wave = threadIdx.x >> 5;
    int lane = threadIdx.x & 31;
    int row  = blockIdx.x * 8 + wave;
    if (row >= n) return;

    float4 hv = ((const float4*)(h + (size_t)row * D_OUT))[lane];
    float4 w1 = ((const float4*)a1w)[lane];
    float4 w2 = ((const float4*)a2w)[lane];
    float s1 = hv.x * w1.x + hv.y * w1.y + hv.z * w1.z + hv.w * w1.w;
    float s2 = hv.x * w2.x + hv.y * w2.y + hv.z * w2.z + hv.w * w2.w;
#pragma unroll
    for (int off = 16; off > 0; off >>= 1) {
        s1 += __shfl_down(s1, off, 32);
        s2 += __shfl_down(s2, off, 32);
    }
    if (lane == 0) {
        f1[row] = s1 + a1b[0];
        f2[row] = s2 + a2b[0];
    }
}

// ---------------------------------------------------------------- kernel 3: edge logits + segment max
__global__ __launch_bounds__(256) void gat_logits_kernel(const int* __restrict__ er,
                                                         const int* __restrict__ ec,
                                                         const float* __restrict__ wts,
                                                         const float* __restrict__ f1,
                                                         const float* __restrict__ f2,
                                                         float* __restrict__ v,
                                                         float* __restrict__ m,
                                                         int E) {
    int e = blockIdx.x * blockDim.x + threadIdx.x;
    if (e >= E) return;
    int r = er[e];
    float x = f1[r] + f2[ec[e]];
    x = (x > 0.0f) ? x : ALPHA * x;   // leaky relu
    x *= wts[e];
    v[e] = x;
    atomicMaxFloat(m + r, x);
}

// ---------------------------------------------------------------- kernel 4: exp + segment sum
__global__ __launch_bounds__(256) void gat_exp_kernel(const int* __restrict__ er,
                                                      const float* __restrict__ m,
                                                      float* __restrict__ v,
                                                      float* __restrict__ denom,
                                                      int E) {
    int e = blockIdx.x * blockDim.x + threadIdx.x;
    if (e >= E) return;
    int r = er[e];
    float ex = expf(v[e] - m[r]);   // m[r] finite: row r has this edge
    v[e] = ex;
    atomicAdd(denom + r, ex);
}

// ---------------------------------------------------------------- kernel 5: SpMM out[row] += coef * h[col]
// One wave per edge; lane handles 4 consecutive features (float4 gather, 4 f32 atomics).
__global__ __launch_bounds__(256) void gat_spmm_kernel(const int* __restrict__ er,
                                                       const int* __restrict__ ec,
                                                       const float* __restrict__ v,
                                                       const float* __restrict__ denom,
                                                       const float* __restrict__ h,
                                                       float* __restrict__ out,
                                                       int E) {
    int gwave = (int)((blockIdx.x * (size_t)blockDim.x + threadIdx.x) >> 5);
    int lane  = threadIdx.x & 31;
    if (gwave >= E) return;
    int r = er[gwave];
    int c = ec[gwave];
    float coef = v[gwave] / fmaxf(denom[r], 1e-38f);
    float4 hv = ((const float4*)(h + (size_t)c * D_OUT))[lane];
    float* op = out + (size_t)r * D_OUT + lane * 4;
    atomicAdd(op + 0, coef * hv.x);
    atomicAdd(op + 1, coef * hv.y);
    atomicAdd(op + 2, coef * hv.z);
    atomicAdd(op + 3, coef * hv.w);
}

// ---------------------------------------------------------------- kernel 6: epilogue relu(out + bias)
__global__ __launch_bounds__(256) void gat_finish_kernel(float* __restrict__ out,
                                                         const float* __restrict__ bias,
                                                         int n) {
    int i = blockIdx.x * blockDim.x + threadIdx.x;
    if (i < n * D_OUT) {
        float x = out[i] + bias[i & (D_OUT - 1)];
        out[i] = fmaxf(x, 0.0f);
    }
}

// ---------------------------------------------------------------- launcher
extern "C" void kernel_launch(void* const* d_in, const int* in_sizes, int n_in,
                              void* d_out, int out_size, void* d_ws, size_t ws_size,
                              hipStream_t stream) {
    (void)n_in; (void)out_size; (void)ws_size;

    const float* seq  = (const float*)d_in[0];
    const int*   er   = (const int*)d_in[1];
    const int*   ec   = (const int*)d_in[2];
    const float* wts  = (const float*)d_in[3];
    const float* W    = (const float*)d_in[4];
    const float* a1w  = (const float*)d_in[5];
    const float* a1b  = (const float*)d_in[6];
    const float* a2w  = (const float*)d_in[7];
    const float* a2b  = (const float*)d_in[8];
    const float* bias = (const float*)d_in[9];
    float* out = (float*)d_out;

    const int N = in_sizes[0] / F_IN;
    const int E = in_sizes[1];

    // workspace partition (256-byte aligned)
    char* ws = (char*)d_ws;
    size_t off = 0;
    auto take = [&](size_t bytes) -> char* {
        char* p = ws + off;
        off = (off + bytes + 255) & ~(size_t)255;
        return p;
    };
    float* h     = (float*)take((size_t)N * D_OUT * sizeof(float));
    float* f1    = (float*)take((size_t)N * sizeof(float));
    float* f2    = (float*)take((size_t)N * sizeof(float));
    float* v     = (float*)take((size_t)E * sizeof(float));
    float* m     = (float*)take((size_t)N * sizeof(float));
    float* denom = (float*)take((size_t)N * sizeof(float));

    // 0: init accumulators
    {
        int total = N * D_OUT;
        gat_init_kernel<<<(total + 255) / 256, 256, 0, stream>>>(out, m, denom, N);
    }
    // 1: h = seq @ W  (WMMA f16 -> f32)
    {
        int blocks = (N + 15) / 16;
        gat_gemm_kernel<<<blocks, 256, 0, stream>>>(seq, W, h, N);
    }
    // 2: f1, f2
    {
        int blocks = (N + 7) / 8;
        gat_f12_kernel<<<blocks, 256, 0, stream>>>(h, a1w, a2w, a1b, a2b, f1, f2, N);
    }
    // 3: logits + segment max
    gat_logits_kernel<<<(E + 255) / 256, 256, 0, stream>>>(er, ec, wts, f1, f2, v, m, E);
    // 4: exp + segment sum
    gat_exp_kernel<<<(E + 255) / 256, 256, 0, stream>>>(er, m, v, denom, E);
    // 5: SpMM (wave per edge)
    {
        long long waves = (long long)E;
        long long blocks = (waves * 32 + 255) / 256;
        gat_spmm_kernel<<<(int)blocks, 256, 0, stream>>>(er, ec, v, denom, h, out, E);
    }
    // 6: relu(out + bias)
    gat_finish_kernel<<<(N * D_OUT + 255) / 256, 256, 0, stream>>>(out, bias, N);
}
